// ConvAlignLoss_85925115724093
// MI455X (gfx1250) — matrix operand order repaired and checked by reading, not compiled, not run on hardware
//
#include <hip/hip_runtime.h>
#include <hip/hip_bf16.h>

typedef __attribute__((ext_vector_type(16))) _Float16 v16h;
typedef __attribute__((ext_vector_type(8)))  _Float16 v8h;
typedef __attribute__((ext_vector_type(8)))  float    v8f;

namespace {
constexpr int kB     = 512;
constexpr int kL1    = 16384;
constexpr int kL2    = 2048;
constexpr int kConv  = 14337;           // L1 - L2 + 1
constexpr int kCC    = 28673;           // 2*kConv - 1
constexpr int kCropStart = 7040;        // (kConv-256)/2

constexpr int kPL1   = 16960;           // padded pred row (f16)
constexpr int kQ1    = kL2 / 32;        // 64 k-chunks
constexpr int kNB1   = (kConv + 511) / 512;   // 29

constexpr int kKcc   = 14368;           // target padded to 449*32
constexpr int kQcc   = kKcc / 32;       // 449
constexpr int kXL    = 43584;           // padded X row (f16)
constexpr int kXPad  = kConv - 1;       // 14336 leading zeros in X
constexpr int kXEnd  = kXPad + kConv - 1;     // 28672: last nonzero index in X
constexpr int kNB2   = (kCC + 511) / 512;     // 57
constexpr int kConvP = 14340;           // conv row stride (f32)
}

__device__ inline v8h load8h_u(const _Float16* p) {
  v8h r;
  __builtin_memcpy(&r, p, 16);   // 2-byte aligned source -> unaligned b128
  return r;
}

// Skewed-Toeplitz correlation tile: computes y[t] = sum_{k<32*Q} x[t+k]*w[k]
// for t = tBase + M + 32*N, M,N in [0,16). xrow points at x+tBase; wrow at w.
// Iterations p in [pLo, pHi) only (caller may clamp out all-zero A windows).
__device__ inline v8f corr_tile(const _Float16* __restrict__ xrow,
                                const _Float16* __restrict__ wrow, int Q,
                                int pLo, int pHi) {
  const int lane = threadIdx.x & 31;
  const int jn   = lane & 15;          // B column / A row (M)
  const int hi   = lane >> 4;
  const int aOff = jn + (hi ? 8 : 0);  // A: lanes<16 hold K{0..7,16..23}, else K{8..15,24..31}
  const _Float16* wl = wrow + 8 * hi;  // B: lanes<16 hold K{0..7,16..23}, else K{8..15,24..31}
  v8f acc = {};
  for (int p = pLo; p < pHi; ++p) {
    v8h a0 = load8h_u(xrow + 32 * p + aOff);
    v8h a1 = load8h_u(xrow + 32 * p + aOff + 16);
    v16h a;
#pragma unroll
    for (int e = 0; e < 8; ++e) { a[e] = a0[e]; a[e + 8] = a1[e]; }
    v16h bm = {};
    const int q = p - jn;
    if ((unsigned)q < (unsigned)Q) {
      v8h b0 = *(const v8h*)(wl + 32 * q);
      v8h b1 = *(const v8h*)(wl + 32 * q + 16);
#pragma unroll
      for (int e = 0; e < 8; ++e) { bm[e] = b0[e]; bm[e + 8] = b1[e]; }
    }
    acc = __builtin_amdgcn_wmma_f32_16x16x32_f16(false, a, false, bm,
                                                 (short)0, acc, false, false);
  }
  return acc;
}

// ---------- staging / conversion kernels ----------
__global__ void k_cvt_pred(const float* __restrict__ src, _Float16* __restrict__ dst) {
  size_t i = (size_t)blockIdx.x * blockDim.x + threadIdx.x;
  if (i >= (size_t)kB * kPL1) return;
  int row = (int)(i / kPL1), col = (int)(i % kPL1);
  float v = (col < kL1) ? src[(size_t)row * kL1 + col] : 0.f;
  dst[i] = (_Float16)v;
}

__global__ void k_cvt_egf(const float* __restrict__ src, _Float16* __restrict__ dst) {
  size_t i = (size_t)blockIdx.x * blockDim.x + threadIdx.x;
  if (i >= (size_t)kB * kL2) return;
  dst[i] = (_Float16)src[i];
}

__global__ void k_cvt_tgt(const float* __restrict__ src, _Float16* __restrict__ dst) {
  size_t i = (size_t)blockIdx.x * blockDim.x + threadIdx.x;
  if (i >= (size_t)kB * kKcc) return;
  int row = (int)(i / kKcc), col = (int)(i % kKcc);
  float v = (col < kConv) ? src[(size_t)row * kConv + col] : 0.f;
  dst[i] = (_Float16)v;
}

__global__ void k_build_x(const float* __restrict__ conv, _Float16* __restrict__ dst) {
  size_t i = (size_t)blockIdx.x * blockDim.x + threadIdx.x;
  if (i >= (size_t)kB * kXL) return;
  int row = (int)(i / kXL), col = (int)(i % kXL);
  float v = 0.f;
  int u = col - kXPad;
  if (u >= 0 && u < kConv) v = conv[(size_t)row * kConvP + u];
  dst[i] = (_Float16)v;
}

// ---------- loss_astf ----------
__global__ void k_mse_astf(const float* __restrict__ a, const float* __restrict__ b,
                           float* __restrict__ acc, int n) {
  __shared__ float sm[256];
  float s = 0.f;
  for (int i = blockIdx.x * blockDim.x + threadIdx.x; i < n;
       i += gridDim.x * blockDim.x) {
    float d = a[i] - b[i];
    s += d * d;
  }
  sm[threadIdx.x] = s;
  __syncthreads();
  for (int o = 128; o > 0; o >>= 1) {
    if ((int)threadIdx.x < o) sm[threadIdx.x] += sm[threadIdx.x + o];
    __syncthreads();
  }
  if (threadIdx.x == 0) atomicAdd(&acc[0], sm[0]);
}

// ---------- grouped conv via WMMA ----------
__global__ __launch_bounds__(256) void k_conv_wmma(
    const _Float16* __restrict__ pred16, const _Float16* __restrict__ egf16,
    float* __restrict__ conv) {
  const int wid = blockIdx.x * (blockDim.x >> 5) + (threadIdx.x >> 5);
  const int tilesPerB = kNB1 * 2;
  if (wid >= kB * tilesPerB) return;   // wave-uniform
  const int b = wid / tilesPerB;
  const int r = wid - b * tilesPerB;
  const int tBase = (r >> 1) * 512 + (r & 1) * 16;
  v8f acc = corr_tile(pred16 + (size_t)b * kPL1 + tBase,
                      egf16 + (size_t)b * kL2, kQ1, 0, kQ1 + 15);
  const int lane = threadIdx.x & 31;
  const int t0 = tBase + 8 * (lane >> 4) + 32 * (lane & 15);
  float* crow = conv + (size_t)b * kConvP;
#pragma unroll
  for (int rr = 0; rr < 8; ++rr) {
    int t = t0 + rr;
    if (t < kConv) crow[t] = acc[rr];
  }
}

// ---------- cross-correlation + argmax via WMMA ----------
__global__ __launch_bounds__(256) void k_cc_wmma(
    const _Float16* __restrict__ x16, const _Float16* __restrict__ tgt16,
    unsigned long long* __restrict__ packed) {
  const int wid = blockIdx.x * (blockDim.x >> 5) + (threadIdx.x >> 5);
  const int tilesPerB = kNB2 * 2;
  if (wid >= kB * tilesPerB) return;   // wave-uniform
  const int b = wid / tilesPerB;
  const int r = wid - b * tilesPerB;
  const int tBase = (r >> 1) * 512 + (r & 1) * 16;

  // Iteration p reads X[tBase+32p .. tBase+32p+46]; X nonzero only on
  // [kXPad, kXEnd].  Clamp the p-loop to windows overlapping that band
  // (wave-uniform; skipped terms are exact zeros).
  int pLo = 0;
  {
    int d = kXPad - 46 - tBase;              // need 32p >= d
    if (d > 0) pLo = (d + 31) >> 5;
  }
  int pHi = kQcc + 15;
  {
    int e = kXEnd - tBase;                   // need 32p <= e
    int lim = (e >= 0) ? (e >> 5) + 1 : 0;
    if (lim < pHi) pHi = lim;
  }

  v8f acc = corr_tile(x16 + (size_t)b * kXL + tBase,
                      tgt16 + (size_t)b * kKcc, kQcc, pLo, pHi);
  const int lane = threadIdx.x & 31;
  const int t0 = tBase + 8 * (lane >> 4) + 32 * (lane & 15);
  unsigned long long best = 0;
#pragma unroll
  for (int rr = 0; rr < 8; ++rr) {
    int t = t0 + rr;
    if (t < kCC) {
      // cc index s: t>=conv_len-1 -> s=t-(conv_len-1); else s=t+conv_len
      int s = (t >= kConv - 1) ? (t - (kConv - 1)) : (t + kConv);
      unsigned u = __float_as_uint(acc[rr]);
      u = (u & 0x80000000u) ? ~u : (u | 0x80000000u);  // order-preserving map
      unsigned long long key =
          ((unsigned long long)u << 32) | (unsigned)(kCC - 1 - s);
      if (key > best) best = key;
    }
  }
#pragma unroll
  for (int off = 16; off > 0; off >>= 1) {
    unsigned long long o = __shfl_down(best, (unsigned)off, 32);
    if (o > best) best = o;
  }
  if (lane == 0 && best) atomicMax(&packed[b], best);
}

// ---------- roll + crop + MSE ----------
__global__ void k_loss_conv(const float* __restrict__ conv,
                            const float* __restrict__ tgt,
                            const unsigned long long* __restrict__ packed,
                            float* __restrict__ acc) {
  const int b = blockIdx.x;
  const int j = threadIdx.x;  // 256
  const unsigned long long pk = packed[b];
  const int s = kCC - 1 - (int)(pk & 0xFFFFFFFFull);
  const int shift = s - (kConv - 1);
  const int jj = kCropStart + j;
  int idx = (jj + shift) % kConv;
  if (idx < 0) idx += kConv;
  float d = conv[(size_t)b * kConvP + idx] - tgt[(size_t)b * kConv + jj];
  __shared__ float sm[256];
  sm[j] = d * d;
  __syncthreads();
  for (int o = 128; o > 0; o >>= 1) {
    if (j < o) sm[j] += sm[j + o];
    __syncthreads();
  }
  if (j == 0) atomicAdd(&acc[1], sm[0]);
}

__global__ void k_finalize(const float* __restrict__ acc, float* __restrict__ out) {
  if (threadIdx.x == 0 && blockIdx.x == 0) {
    float la = acc[0] / (float)((size_t)kB * kL1);
    float lc = acc[1] / (float)(kB * 256);
    out[0] = la + lc;
    out[1] = la;
    out[2] = lc;
  }
}

extern "C" void kernel_launch(void* const* d_in, const int* in_sizes, int n_in,
                              void* d_out, int out_size, void* d_ws, size_t ws_size,
                              hipStream_t stream) {
  const float* pred = (const float*)d_in[0];
  const float* truth = (const float*)d_in[1];
  const float* egf = (const float*)d_in[2];
  const float* tgt = (const float*)d_in[3];

  char* ws = (char*)d_ws;
  size_t o = 0;
  _Float16* pred16 = (_Float16*)(ws + o); o += (size_t)kB * kPL1 * 2;
  _Float16* egf16  = (_Float16*)(ws + o); o += (size_t)kB * kL2 * 2;
  _Float16* tgt16  = (_Float16*)(ws + o); o += (size_t)kB * kKcc * 2;
  float*    conv   = (float*)(ws + o);    o += (size_t)kB * kConvP * 4;
  _Float16* x16    = (_Float16*)(ws + o); o += (size_t)kB * kXL * 2;
  unsigned long long* packed = (unsigned long long*)(ws + o); o += (size_t)kB * 8;
  float*    acc    = (float*)(ws + o);    o += 256;

  // zero argmax keys + loss accumulators (contiguous)
  hipMemsetAsync(packed, 0, (size_t)kB * 8 + 256, stream);

  const int T = 256;
  k_cvt_pred<<<(int)(((size_t)kB * kPL1 + T - 1) / T), T, 0, stream>>>(pred, pred16);
  k_cvt_egf<<<(int)(((size_t)kB * kL2 + T - 1) / T), T, 0, stream>>>(egf, egf16);
  k_cvt_tgt<<<(int)(((size_t)kB * kKcc + T - 1) / T), T, 0, stream>>>(tgt, tgt16);
  k_mse_astf<<<2048, T, 0, stream>>>(pred, truth, acc, kB * kL1);

  const int waves1 = kB * kNB1 * 2;                  // 29696
  k_conv_wmma<<<(waves1 + 7) / 8, T, 0, stream>>>(pred16, egf16, conv);

  k_build_x<<<(int)(((size_t)kB * kXL + T - 1) / T), T, 0, stream>>>(conv, x16);

  const int waves2 = kB * kNB2 * 2;                  // 58368
  k_cc_wmma<<<(waves2 + 7) / 8, T, 0, stream>>>(x16, tgt16, packed);

  k_loss_conv<<<kB, T, 0, stream>>>(conv, tgt, packed, acc);
  k_finalize<<<1, 32, 0, stream>>>(acc, (float*)d_out);
}
